// MultiHeadedAttention_42820823941268
// MI455X (gfx1250) — matrix operand order repaired and checked by reading, not compiled, not run on hardware
//
#include <hip/hip_runtime.h>
#include <hip/hip_bf16.h>

typedef __attribute__((ext_vector_type(8)))  float  v8f;
typedef __attribute__((ext_vector_type(8)))  __bf16 v8bf;
typedef __attribute__((ext_vector_type(16))) __bf16 v16bf;
typedef __attribute__((ext_vector_type(4)))  unsigned int tdm_v4u;
typedef __attribute__((ext_vector_type(8)))  int          tdm_v8i;
typedef __attribute__((ext_vector_type(4)))  int          tdm_v4i;

constexpr int BB = 2, SS = 2048, DD = 1024, HH = 16, DK = 64;
constexpr int MM = BB * SS;   // 4096 rows for projection GEMMs

// ---------------------------------------------------------------- helpers
__device__ __forceinline__ v16bf pack_bf(v8bf lo, v8bf hi) {
  v16bf r;
#pragma unroll
  for (int i = 0; i < 8; ++i) { r[i] = lo[i]; r[i + 8] = hi[i]; }
  return r;
}
__device__ __forceinline__ v16bf pack_f32(v8f lo, v8f hi) {
  v16bf r;
#pragma unroll
  for (int i = 0; i < 8; ++i) { r[i] = (__bf16)lo[i]; r[i + 8] = (__bf16)hi[i]; }
  return r;
}
__device__ __forceinline__ v8f wmma_bf16(v16bf a, v16bf b, v8f c) {
  // D = A(16x32 bf16) * B(32x16 bf16) + C(16x16 f32)
  return __builtin_amdgcn_wmma_f32_16x16x32_bf16(false, a, false, b, (short)0, c,
                                                 false, false);
}

// ------------------------------------------------- TDM 2D tile load -> LDS
// Packs a D# (group0 + group1) for a 2D bf16 tile: tile_h rows of tile_w
// contiguous elements, row stride `stride` elements; dest LDS is contiguous.
__device__ __forceinline__ void tdm_load_2d(__bf16* lptr, const __bf16* gptr,
                                            unsigned tile_w, unsigned tile_h,
                                            unsigned stride) {
#if __has_builtin(__builtin_amdgcn_tensor_load_to_lds)
  const unsigned long long ga = (unsigned long long)(size_t)gptr;
  const unsigned lds_off = (unsigned)(size_t)lptr;   // flat->LDS: addr[31:0]
  tdm_v4u g0;
  g0[0] = 1u;                                        // count=1, user descriptor
  g0[1] = lds_off;                                   // lds_addr (bytes)
  g0[2] = (unsigned)(ga & 0xFFFFFFFFu);              // global_addr[31:0]
  g0[3] = (unsigned)((ga >> 32) & 0x1FFFFFFu) | (2u << 30);  // [56:32] | type=2
  tdm_v8i g1;
  g1[0] = (int)(1u << 16);                           // wg_mask=0, data_size=1 (2B)
  g1[1] = (int)((tile_w & 0xFFFFu) << 16);           // tensor_dim0[15:0]
  g1[2] = (int)((tile_w >> 16) | ((tile_h & 0xFFFFu) << 16));   // dim0 hi | dim1 lo
  g1[3] = (int)((tile_h >> 16) | (tile_w << 16));    // dim1 hi | tile_dim0
  g1[4] = (int)tile_h;                               // tile_dim1, tile_dim2=0
  g1[5] = (int)stride;                               // tensor_dim0_stride[31:0]
  g1[6] = 0;                                         // stride0 hi | stride1 lo
  g1[7] = 0;
  tdm_v4i z4 = {0, 0, 0, 0};
  tdm_v8i z8 = {0, 0, 0, 0, 0, 0, 0, 0};
  __builtin_amdgcn_tensor_load_to_lds(g0, g1, z4, z4, z8, 0);
#else
  // Fallback: cooperative copy by the calling wave (compile-safe path)
  const int lane = threadIdx.x & 31;
  for (unsigned r = 0; r < tile_h; ++r)
    for (unsigned c = (unsigned)lane * 8u; c < tile_w; c += 256u)
      *(v8bf*)(lptr + (size_t)r * tile_w + c) =
          *(const v8bf*)(gptr + (size_t)r * stride + c);
#endif
}
template <int N>
__device__ __forceinline__ void tdm_wait() {
#if __has_builtin(__builtin_amdgcn_s_wait_tensorcnt)
  __builtin_amdgcn_s_wait_tensorcnt(N);   // immediate required
#else
  __builtin_amdgcn_wave_barrier();
#endif
}

// ---------------------------------------------------------------- fp32 -> bf16
__global__ void cvt_f32_to_bf16(const float* __restrict__ src,
                                __bf16* __restrict__ dst, int n) {
  int i = blockIdx.x * blockDim.x + threadIdx.x;
  if (i < n) dst[i] = (__bf16)src[i];
}

// ---------------------------------------------------------------- projections
// y[m,n] = sum_k X[m,k] * W[n,k]   (X fp32 [M,DD], W bf16 [DD,DD] row-major)
// MODE 0: out bf16 row-major [M,DD]           (Q, K)
// MODE 1: out bf16 head-transposed [B,H,DK,S] (V, so PV B-frags are contiguous)
template <int MODE>
__global__ __launch_bounds__(256) void gemm_x_wT(const float* __restrict__ X,
                                                 const __bf16* __restrict__ W,
                                                 __bf16* __restrict__ out) {
  const int lane = threadIdx.x & 31;
  const int wave = threadIdx.x >> 5;
  const int half = lane >> 4;       // 0: lanes 0-15, 1: lanes 16-31
  const int lrow = lane & 15;
  const int m0 = blockIdx.y * 64 + (wave & 3) * 16;
  const int n0 = blockIdx.x * 32 + (wave >> 2) * 16;

  const float*  ap = X + (size_t)(m0 + lrow) * DD + half * 16;  // A row, K contiguous 16 per half
  const __bf16* bp = W + (size_t)(n0 + lrow) * DD + half * 16;  // B col = W row

  v8f acc;
#pragma unroll
  for (int j = 0; j < 8; ++j) acc[j] = 0.0f;

#pragma unroll 4
  for (int kk = 0; kk < DD; kk += 32) {
    v8f  a0 = *(const v8f*)(ap + kk);
    v8f  a1 = *(const v8f*)(ap + kk + 8);
    v8bf b0 = *(const v8bf*)(bp + kk);
    v8bf b1 = *(const v8bf*)(bp + kk + 8);
    __builtin_prefetch(ap + kk + 128, 0, 3);
    __builtin_prefetch(bp + kk + 128, 0, 3);
    acc = wmma_bf16(pack_f32(a0, a1), pack_bf(b0, b1), acc);
  }

  if (MODE == 0) {
#pragma unroll
    for (int j = 0; j < 8; ++j) {
      int mrow = m0 + j + half * 8;
      out[(size_t)mrow * DD + n0 + lrow] = (__bf16)acc[j];
    }
  } else {
    int ncol = n0 + lrow;
    int hh = ncol / DK, dd2 = ncol % DK;
#pragma unroll
    for (int j = 0; j < 8; ++j) {
      int mrow = m0 + j + half * 8;
      int bb = mrow / SS, ss = mrow % SS;
      out[(((size_t)bb * HH + hh) * DK + dd2) * SS + ss] = (__bf16)acc[j];
    }
  }
}

// ---------------------------------------------------------------- flash attention
// One wave per (b, h, 16-row q tile); K/V chunks (32 keys) are DMA'd to LDS by
// the Tensor Data Mover, double-buffered, and consumed as WMMA B-fragments.
__global__ __launch_bounds__(128) void flash_attn(const __bf16* __restrict__ Qb,
                                                  const __bf16* __restrict__ Kb,
                                                  const __bf16* __restrict__ Vt,
                                                  const int* __restrict__ mask,
                                                  __bf16* __restrict__ Ctx) {
  __shared__ __align__(16) __bf16 sK[4][2][32][64];  // keys x dims, 32 KB
  __shared__ __align__(16) __bf16 sV[4][2][64][32];  // dims x keys, 32 KB
  __shared__ __align__(16) __bf16 sP[4][16][32];     // P re-swizzle,  4 KB

  const int wave = threadIdx.x >> 5;
  const int lane = threadIdx.x & 31;
  const int half = lane >> 4;
  const int lrow = lane & 15;

  const int QT = SS / 16;                       // 128 q-tiles per head
  const int tile = blockIdx.x * 4 + wave;       // 0 .. B*H*QT-1
  const int q0 = (tile % QT) * 16;
  const int h  = (tile / QT) % HH;
  const int b  = tile / (QT * HH);

  const __bf16* kbase = Kb + (size_t)(b * SS) * DD + h * DK;        // + s0*DD
  const __bf16* vbase = Vt + (size_t)(b * HH + h) * DK * SS;        // + s0

  // Q A-fragments (16x64 split into two 16x32 K-chunks)
  v16bf qa[2];
  {
    const __bf16* qp = Qb + (size_t)(b * SS + q0 + lrow) * DD + h * DK + half * 16;
#pragma unroll
    for (int dk = 0; dk < 2; ++dk) {
      v8bf lo = *(const v8bf*)(qp + dk * 32);
      v8bf hi = *(const v8bf*)(qp + dk * 32 + 8);
      qa[dk] = pack_bf(lo, hi);
    }
  }

  v8f o[4];                                     // O tile 16x64 (4 N-tiles), fp32
  float mi[8], li[8];
#pragma unroll
  for (int t = 0; t < 4; ++t)
#pragma unroll
    for (int j = 0; j < 8; ++j) o[t][j] = 0.0f;
#pragma unroll
  for (int j = 0; j < 8; ++j) { mi[j] = -1e30f; li[j] = 0.0f; }

  // Prologue: TDM-stage chunk 0 into buffer 0.
  tdm_load_2d(&sK[wave][0][0][0], kbase, DK, 32, DD);
  tdm_load_2d(&sV[wave][0][0][0], vbase, 32, DK, SS);

  for (int s0 = 0, it = 0; s0 < SS; s0 += 32, ++it) {
    const int buf = it & 1;
    if (s0 + 32 < SS) {
      // Prefetch next chunk into the other buffer while we compute this one.
      tdm_load_2d(&sK[wave][buf ^ 1][0][0], kbase + (size_t)(s0 + 32) * DD, DK, 32, DD);
      tdm_load_2d(&sV[wave][buf ^ 1][0][0], vbase + (s0 + 32), 32, DK, SS);
      tdm_wait<2>();   // TDM is in-order per wave: <=2 means current chunk landed
    } else {
      tdm_wait<0>();
    }

    // ---- scores S = Q K^T * 1/8, masked, two 16x16 column tiles (from LDS)
    v8f sc[2];
#pragma unroll
    for (int nt = 0; nt < 2; ++nt) {
#pragma unroll
      for (int j = 0; j < 8; ++j) sc[nt][j] = 0.0f;
      const __bf16* kp = &sK[wave][buf][nt * 16 + lrow][half * 16];
#pragma unroll
      for (int dk = 0; dk < 2; ++dk) {
        v8bf k0 = *(const v8bf*)(kp + dk * 32);
        v8bf k1 = *(const v8bf*)(kp + dk * 32 + 8);
        sc[nt] = wmma_bf16(qa[dk], pack_bf(k0, k1), sc[nt]);
      }
      const int scol = s0 + nt * 16 + lrow;     // this lane's key column
      const int* mp = mask + (size_t)b * SS * SS + scol;
#pragma unroll
      for (int j = 0; j < 8; ++j) {
        int mrow = q0 + j + half * 8;
        int mv = mp[(size_t)mrow * SS];
        float sv = sc[nt][j] * 0.125f;          // 1/sqrt(DK)
        sc[nt][j] = mv ? sv : -1e9f;
      }
    }

    // ---- online softmax per row (16-lane xor-shuffle row reductions)
#pragma unroll
    for (int j = 0; j < 8; ++j) {
      float tmax = fmaxf(sc[0][j], sc[1][j]);
#pragma unroll
      for (int off = 8; off > 0; off >>= 1)
        tmax = fmaxf(tmax, __shfl_xor(tmax, off, 32));
      float nm = fmaxf(mi[j], tmax);
      float p0 = __expf(sc[0][j] - nm);
      float p1 = __expf(sc[1][j] - nm);
      float rs = p0 + p1;
#pragma unroll
      for (int off = 8; off > 0; off >>= 1)
        rs += __shfl_xor(rs, off, 32);
      float corr = __expf(mi[j] - nm);
      li[j] = li[j] * corr + rs;
      mi[j] = nm;
      o[0][j] *= corr; o[1][j] *= corr; o[2][j] *= corr; o[3][j] *= corr;
      sP[wave][j + half * 8][lrow]      = (__bf16)p0;
      sP[wave][j + half * 8][lrow + 16] = (__bf16)p1;
    }
    __builtin_amdgcn_wave_barrier();            // C/D-layout -> A-layout bounce

    // ---- O += P (16x32) @ V-chunk (32x64), V fed from LDS
    v8bf plo = *(const v8bf*)&sP[wave][lrow][half * 16];
    v8bf phi = *(const v8bf*)&sP[wave][lrow][half * 16 + 8];
    v16bf pa = pack_bf(plo, phi);
    __builtin_amdgcn_wave_barrier();
#pragma unroll
    for (int t = 0; t < 4; ++t) {
      const __bf16* vp = &sV[wave][buf][t * 16 + lrow][half * 16];
      v8bf v0 = *(const v8bf*)vp;
      v8bf v1 = *(const v8bf*)(vp + 8);
      o[t] = wmma_bf16(pa, pack_bf(v0, v1), o[t]);
    }
  }

  // ---- normalize and store context (merged heads, bf16 row-major [B,S,D])
#pragma unroll
  for (int t = 0; t < 4; ++t) {
#pragma unroll
    for (int j = 0; j < 8; ++j) {
      int mrow = q0 + j + half * 8;
      int col  = h * DK + t * 16 + lrow;
      Ctx[(size_t)(b * SS + mrow) * DD + col] = (__bf16)(o[t][j] / li[j]);
    }
  }
}

// ---------------------------------------------------------------- out = ctx @ Wo^T (fp32 out)
__global__ __launch_bounds__(256) void gemm_ctx_woT(const __bf16* __restrict__ X,
                                                    const __bf16* __restrict__ W,
                                                    float* __restrict__ out) {
  const int lane = threadIdx.x & 31;
  const int wave = threadIdx.x >> 5;
  const int half = lane >> 4;
  const int lrow = lane & 15;
  const int m0 = blockIdx.y * 64 + (wave & 3) * 16;
  const int n0 = blockIdx.x * 32 + (wave >> 2) * 16;

  const __bf16* ap = X + (size_t)(m0 + lrow) * DD + half * 16;
  const __bf16* bp = W + (size_t)(n0 + lrow) * DD + half * 16;

  v8f acc;
#pragma unroll
  for (int j = 0; j < 8; ++j) acc[j] = 0.0f;

#pragma unroll 4
  for (int kk = 0; kk < DD; kk += 32) {
    v8bf a0 = *(const v8bf*)(ap + kk);
    v8bf a1 = *(const v8bf*)(ap + kk + 8);
    v8bf b0 = *(const v8bf*)(bp + kk);
    v8bf b1 = *(const v8bf*)(bp + kk + 8);
    __builtin_prefetch(ap + kk + 128, 0, 3);
    __builtin_prefetch(bp + kk + 128, 0, 3);
    acc = wmma_bf16(pack_bf(a0, a1), pack_bf(b0, b1), acc);
  }
#pragma unroll
  for (int j = 0; j < 8; ++j) {
    int mrow = m0 + j + half * 8;
    out[(size_t)mrow * DD + n0 + lrow] = acc[j];
  }
}

// ---------------------------------------------------------------- launcher
extern "C" void kernel_launch(void* const* d_in, const int* in_sizes, int n_in,
                              void* d_out, int out_size, void* d_ws, size_t ws_size,
                              hipStream_t stream) {
  (void)in_sizes; (void)n_in; (void)out_size; (void)ws_size;

  const float* q   = (const float*)d_in[0];
  const float* k   = (const float*)d_in[1];
  const float* v   = (const float*)d_in[2];
  const int*   msk = (const int*)d_in[3];
  const float* Wq  = (const float*)d_in[4];
  const float* Wk  = (const float*)d_in[5];
  const float* Wv  = (const float*)d_in[6];
  const float* Wo  = (const float*)d_in[7];
  float* out = (float*)d_out;

  char* ws = (char*)d_ws;
  const size_t MB = 1ull << 20;
  __bf16* Qb  = (__bf16*)(ws + 0 * MB);    // [B,S,D]       8 MB
  __bf16* Kbm = (__bf16*)(ws + 8 * MB);    // [B,S,D]       8 MB
  __bf16* Vt  = (__bf16*)(ws + 16 * MB);   // [B,H,DK,S]    8 MB
  __bf16* Ctx = (__bf16*)(ws + 24 * MB);   // [B,S,D]       8 MB
  __bf16* Wqb = (__bf16*)(ws + 32 * MB);   // 2 MB each
  __bf16* Wkb = (__bf16*)(ws + 34 * MB);
  __bf16* Wvb = (__bf16*)(ws + 36 * MB);
  __bf16* Wob = (__bf16*)(ws + 38 * MB);

  const int nw = DD * DD;
  dim3 cb(256), cg((nw + 255) / 256);
  cvt_f32_to_bf16<<<cg, cb, 0, stream>>>(Wq, Wqb, nw);
  cvt_f32_to_bf16<<<cg, cb, 0, stream>>>(Wk, Wkb, nw);
  cvt_f32_to_bf16<<<cg, cb, 0, stream>>>(Wv, Wvb, nw);
  cvt_f32_to_bf16<<<cg, cb, 0, stream>>>(Wo, Wob, nw);

  dim3 gg(DD / 32, MM / 64), gb(256);
  gemm_x_wT<0><<<gg, gb, 0, stream>>>(q, Wqb, Qb);
  gemm_x_wT<0><<<gg, gb, 0, stream>>>(k, Wkb, Kbm);
  gemm_x_wT<1><<<gg, gb, 0, stream>>>(v, Wvb, Vt);

  flash_attn<<<(BB * HH * (SS / 16)) / 4, 128, 0, stream>>>(Qb, Kbm, Vt, msk, Ctx);

  gemm_ctx_woT<<<gg, gb, 0, stream>>>(Ctx, Wob, out);
}
